// Joint2D3DEGNNModel_44521630991106
// MI455X (gfx1250) — compile-verified, matrix operand order checked by reading, not run on hardware
//
#include <hip/hip_runtime.h>
#include <hip/hip_bf16.h>

// ---------------------------------------------------------------------------
// Joint 2D/3D EGNN forward for MI455X (gfx1250, wave32, WMMA f16->f32).
// V_WMMA_F32_16X16X32_F16 tiled GEMMs: 128x128x32 block tiles, 8 waves,
// each wave 64x32 = 4x2 WMMA tiles. A staged in LDS (vectorized f32->f16,
// row-clamped instead of guarded: garbage rows only feed unstored C rows).
// B read directly from a pre-converted f16 transposed weight copy [N][K]
// (L2-resident) as aligned 16B fragment chunks. Edge MLPs run in 128K-edge
// chunks with f16 activations; e1 is split into node GEMMs Ha/Hb and the
// per-edge gather + silu is fused into the A-tile load of the e2 GEMM.
// Epilogue: block-uniform full/tail split, bias always valid (zero buffer).
// ---------------------------------------------------------------------------

#define HID 256
#define NN  40000
#define E2N 120000
#define E3N 640000
#define NL  4
#define CHUNK 131072   // multiple of 128; E3N tail = 115712 also mult of 128

typedef _Float16 v16h __attribute__((ext_vector_type(16)));
typedef float    v8f  __attribute__((ext_vector_type(8)));

union FragAB { v16h v; uint4 q[2]; };
union FragC  { v8f  v; float f[8]; };
union H8     { uint4 q; _Float16 h[8]; };

__device__ __forceinline__ float siluf(float x)    { return x / (1.f + __expf(-x)); }
__device__ __forceinline__ float sigmoidf_(float x){ return 1.f / (1.f + __expf(-x)); }

enum { SRC_F32 = 0, SRC_CONCAT = 1, SRC_F16 = 2, SRC_GATHER = 3 };
enum { EPI_NONE = 0, EPI_SILU = 1 };
enum { OUT_F32 = 0, OUT_F16 = 1 };

#define BM 128
#define BN 128
#define BK 32
#define LDA 40   // padded LDS row stride in halves; 80B rows keep 16B alignment

// W[K][256] f32 -> Wt[n][K] f16 (transposed, fragment-friendly)
__global__ void __launch_bounds__(256)
k_cvtW(const float* __restrict__ W, _Float16* __restrict__ Wt, int K) {
  int idx = blockIdx.x * 256 + threadIdx.x;   // over K*256 elements
  int k = idx >> 8, n = idx & 255;
  Wt[(size_t)n * K + k] = (_Float16)W[idx];
}

// ------------------------------ WMMA GEMM ----------------------------------
// C[M,256] = epi(A[M,K] @ W[K,256] + bias); Wt is f16 transposed [256][K].
template<int ASRC, int EPI, int OUTF>
__global__ void __launch_bounds__(256)
k_gemm(const float* __restrict__ A0, const float* __restrict__ A1,
       const _Float16* __restrict__ A16,
       const int* __restrict__ gRow, const int* __restrict__ gCol,
       const float* __restrict__ Ha, const float* __restrict__ Hb,
       const float* __restrict__ p3old,
       const float* __restrict__ wr, const float* __restrict__ b1,
       const _Float16* __restrict__ Wt, const float* __restrict__ bias,
       float* __restrict__ Cf, _Float16* __restrict__ Ch,
       int M, int K, int ebase)
{
  __shared__ _Float16 As[BM * LDA];
  __shared__ int   sRi[BM];
  __shared__ int   sCi[BM];
  __shared__ float sRn[BM];

  const int tid   = threadIdx.x;
  const int mBase = blockIdx.x * BM;
  const int nBase = blockIdx.y * BN;
  const int wid   = tid >> 5;
  const int lane  = tid & 31;
  const int waveM = (wid >> 2) * 64;   // 2 wave-rows of 64
  const int waveN = (wid & 3) * 32;    // 4 wave-cols of 32
  const int rA    = tid >> 2;          // staging row (plus rA+64)
  const int c0    = (tid & 3) * 8;     // staging 8-col group

  if (ASRC == SRC_GATHER) {
    if (tid < BM) {
      int e  = ebase + mBase + tid;
      int ri = gRow[e], ci = gCol[e];
      float dx = p3old[ri*3+0] - p3old[ci*3+0];
      float dy = p3old[ri*3+1] - p3old[ci*3+1];
      float dz = p3old[ri*3+2] - p3old[ci*3+2];
      sRi[tid] = ri; sCi[tid] = ci;
      sRn[tid] = fmaxf(sqrtf(dx*dx + dy*dy + dz*dz), 1e-8f);
    }
    __syncthreads();
  }

  FragC acc[4][2];
#pragma unroll
  for (int i = 0; i < 4; ++i)
#pragma unroll
    for (int j = 0; j < 2; ++j)
#pragma unroll
      for (int r = 0; r < 8; ++r) acc[i][j].f[r] = 0.f;

  const int hi = (lane & 16) ? 8 : 0;
  const int lm = lane & 15;

  const int kSteps = K / BK;
  for (int ks = 0; ks < kSteps; ++ks) {
    const int kBase = ks * BK;

    // ---- stage A tile (128x32 f16), branchless: clamp row, vector loads ----
#pragma unroll
    for (int rr = 0; rr < 2; ++rr) {
      const int r  = rA + rr * 64;
      H8 h8;
      if (ASRC == SRC_F16) {
        int gm = mBase + r; gm = (gm < M) ? gm : (M - 1);
        h8.q = *(const uint4*)(A16 + (size_t)gm * HID + kBase + c0);
      } else if (ASRC == SRC_F32 || ASRC == SRC_CONCAT) {
        int gm = mBase + r; gm = (gm < M) ? gm : (M - 1);
        const float* s;
        if (ASRC == SRC_F32) {
          s = A0 + (size_t)gm * K + kBase + c0;
        } else {
          int cg = kBase + c0;   // 8-col group never straddles the 256 split
          s = (cg < HID) ? (A0 + (size_t)gm * HID + cg)
                         : (A1 + (size_t)gm * HID + (cg - HID));
        }
        float4 x0 = *(const float4*)s;
        float4 x1 = *(const float4*)(s + 4);
        float f[8] = {x0.x, x0.y, x0.z, x0.w, x1.x, x1.y, x1.z, x1.w};
#pragma unroll
        for (int u = 0; u < 8; ++u) h8.h[u] = (_Float16)f[u];
      } else { // SRC_GATHER: silu(Ha[row] + Hb[col] + rn*wr + b1)
        const int cg = kBase + c0;
        const float rn = sRn[r];
        const float* pa = Ha + (size_t)sRi[r] * HID + cg;
        const float* pb = Hb + (size_t)sCi[r] * HID + cg;
        float4 a0 = *(const float4*)pa,        a1 = *(const float4*)(pa + 4);
        float4 q0 = *(const float4*)pb,        q1 = *(const float4*)(pb + 4);
        float4 w0 = *(const float4*)(wr + cg), w1 = *(const float4*)(wr + cg + 4);
        float4 g0 = *(const float4*)(b1 + cg), g1 = *(const float4*)(b1 + cg + 4);
        float fa[8] = {a0.x, a0.y, a0.z, a0.w, a1.x, a1.y, a1.z, a1.w};
        float fb[8] = {q0.x, q0.y, q0.z, q0.w, q1.x, q1.y, q1.z, q1.w};
        float fw[8] = {w0.x, w0.y, w0.z, w0.w, w1.x, w1.y, w1.z, w1.w};
        float fg[8] = {g0.x, g0.y, g0.z, g0.w, g1.x, g1.y, g1.z, g1.w};
#pragma unroll
        for (int u = 0; u < 8; ++u)
          h8.h[u] = (_Float16)siluf(fa[u] + fb[u] + rn * fw[u] + fg[u]);
      }
      *(uint4*)(As + r * LDA + c0) = h8.q;
    }
    __syncthreads();

    // ---- fragments: A from LDS (2x16B), B from global Wt (2x16B) ----
    FragAB af[4], bf[2];
#pragma unroll
    for (int i = 0; i < 4; ++i) {
      const _Float16* p = &As[(waveM + i * 16 + lm) * LDA + hi];
      af[i].q[0] = *(const uint4*)p;
      af[i].q[1] = *(const uint4*)(p + 16);
    }
#pragma unroll
    for (int j2 = 0; j2 < 2; ++j2) {
      const _Float16* p = Wt + (size_t)(nBase + waveN + j2 * 16 + lm) * K + kBase + hi;
      bf[j2].q[0] = *(const uint4*)p;
      bf[j2].q[1] = *(const uint4*)(p + 16);
    }
#pragma unroll
    for (int i = 0; i < 4; ++i)
#pragma unroll
      for (int j = 0; j < 2; ++j)
        acc[i][j].v = __builtin_amdgcn_wmma_f32_16x16x32_f16(
            false, af[i].v, false, bf[j].v, (short)0, acc[i][j].v, false, false);
    __syncthreads();
  }

  // ---- epilogue: bias + activation + store (C/D: M = r + 8*(lane>=16)) ----
  const int rowHlf = (lane >> 4) * 8;
  if (mBase + BM <= M) {
    // full block: unguarded stores (all edge GEMMs + all but last node block)
#pragma unroll
    for (int i = 0; i < 4; ++i) {
#pragma unroll
      for (int j = 0; j < 2; ++j) {
        int gcol = nBase + waveN + j * 16 + lm;
        float bv = bias[gcol];
#pragma unroll
        for (int r = 0; r < 8; ++r) {
          int grow = mBase + waveM + i * 16 + rowHlf + r;
          float v = acc[i][j].f[r] + bv;
          if (EPI == EPI_SILU) v = siluf(v);
          if (OUTF == OUT_F32) Cf[(size_t)grow * HID + gcol] = v;
          else                 Ch[(size_t)grow * HID + gcol] = (_Float16)v;
        }
      }
    }
  } else {
#pragma unroll
    for (int i = 0; i < 4; ++i) {
#pragma unroll
      for (int j = 0; j < 2; ++j) {
        int gcol = nBase + waveN + j * 16 + lm;
        float bv = bias[gcol];
#pragma unroll
        for (int r = 0; r < 8; ++r) {
          int grow = mBase + waveM + i * 16 + rowHlf + r;
          if (grow < M) {
            float v = acc[i][j].f[r] + bv;
            if (EPI == EPI_SILU) v = siluf(v);
            if (OUTF == OUT_F32) Cf[(size_t)grow * HID + gcol] = v;
            else                 Ch[(size_t)grow * HID + gcol] = (_Float16)v;
          }
        }
      }
    }
  }
}

// ------------------------------ helpers ------------------------------------
__global__ void __launch_bounds__(256)
k_zero(float* __restrict__ p, size_t n) {
  size_t i = (size_t)blockIdx.x * 256 + threadIdx.x;
  if (i < n) p[i] = 0.f;
}

__global__ void __launch_bounds__(256)
k_embed(const float* __restrict__ x, const float* __restrict__ W,
        const float* __restrict__ b, float* __restrict__ h) {
  int i = blockIdx.x, j = threadIdx.x;
  float acc = b[j];
#pragma unroll
  for (int k = 0; k < 8; ++k) acc += x[(size_t)i * 8 + k] * W[k * HID + j];
  h[(size_t)i * HID + j] = acc;
}

__global__ void __launch_bounds__(256)
k_scatter2d(const float* __restrict__ xt, float* __restrict__ out,
            const int* __restrict__ ei2, const float* __restrict__ ea,
            const float* __restrict__ eW, const float* __restrict__ eb) {
  int e = blockIdx.x, k = threadIdx.x;
  int ri = ei2[e], ci = ei2[E2N + e];
  float v = xt[(size_t)ci * HID + k] + ea[e] * eW[k] + eb[k];
  unsafeAtomicAdd(&out[(size_t)ri * HID + k], v);
}

// MODE 0: h += LN(silu(x))   (2D layers)     MODE 1: h = LN(h + x)  (EGNN)
template<int MODE>
__global__ void __launch_bounds__(256)
k_layernorm(const float* __restrict__ x, float* __restrict__ h,
            const float* __restrict__ g, const float* __restrict__ b) {
  int row = blockIdx.x, k = threadIdx.x;
  float v = x[(size_t)row * HID + k];
  if (MODE == 0) v = siluf(v);
  else           v = h[(size_t)row * HID + k] + v;
  __shared__ float s1[256], s2[256];
  s1[k] = v; s2[k] = v * v;
  __syncthreads();
  for (int off = 128; off > 0; off >>= 1) {
    if (k < off) { s1[k] += s1[k + off]; s2[k] += s2[k + off]; }
    __syncthreads();
  }
  float mu  = s1[0] * (1.f / 256.f);
  float var = s2[0] * (1.f / 256.f) - mu * mu;
  float y = (v - mu) * rsqrtf(var + 1e-5f) * g[k] + b[k];
  if (MODE == 0) h[(size_t)row * HID + k] += y;
  else           h[(size_t)row * HID + k]  = y;
}

// wave-per-edge: cd = (tanh?)(t@wc2 + b); scatter cd*radial/rn into p3new
__global__ void __launch_bounds__(256)
k_cdpos(const _Float16* __restrict__ t16, const float* __restrict__ wc2,
        const float* __restrict__ bc2, const int* __restrict__ gRow,
        const int* __restrict__ gCol, const float* __restrict__ p3old,
        float* __restrict__ p3new, int base, int cs, int lastTanh) {
  int lw = blockIdx.x * 8 + (threadIdx.x >> 5);
  if (lw >= cs) return;
  int lane = threadIdx.x & 31;
  const _Float16* tr = t16 + (size_t)lw * HID;
  float p = 0.f;
#pragma unroll
  for (int i = 0; i < 8; ++i) { int k = lane + 32 * i; p += (float)tr[k] * wc2[k]; }
#pragma unroll
  for (int mm = 16; mm >= 1; mm >>= 1) p += __shfl_xor(p, mm, 32);
  if (lane == 0) {
    float cd = p + bc2[0];
    if (lastTanh) cd = tanhf(cd);
    int e = base + lw;
    int ri = gRow[e], ci = gCol[e];
    float dx = p3old[ri*3+0] - p3old[ci*3+0];
    float dy = p3old[ri*3+1] - p3old[ci*3+1];
    float dz = p3old[ri*3+2] - p3old[ci*3+2];
    float rn = fmaxf(sqrtf(dx*dx + dy*dy + dz*dz), 1e-8f);
    float s = cd / rn;
    unsafeAtomicAdd(&p3new[ri*3+0], s * dx);
    unsafeAtomicAdd(&p3new[ri*3+1], s * dy);
    unsafeAtomicAdd(&p3new[ri*3+2], s * dz);
  }
}

// block-per-edge: optional sigmoid attention gate, then scatter-add m -> agg
template<int HASATT>
__global__ void __launch_bounds__(256)
k_aggatt(const _Float16* __restrict__ m16, const float* __restrict__ attW,
         const float* __restrict__ attB, const int* __restrict__ gRow,
         float* __restrict__ agg, int base) {
  int le = blockIdx.x, k = threadIdx.x;
  float mv = (float)m16[(size_t)le * HID + k];
  float a = 1.f;
  if (HASATT) {
    __shared__ float red[256];
    red[k] = mv * attW[k];
    __syncthreads();
    for (int off = 128; off > 0; off >>= 1) {
      if (k < off) red[k] += red[k + off];
      __syncthreads();
    }
    a = sigmoidf_(red[0] + attB[0]);
  }
  int ri = gRow[base + le];
  unsafeAtomicAdd(&agg[(size_t)ri * HID + k], a * mv);
}

// wave-per-row small-output head: out[r,:OD] = A[r,:KD]@W + b (+ addb)
template<int KD, int OD, int GATHER, int ADDB>
__global__ void __launch_bounds__(256)
k_smallout(const float* __restrict__ A, const int* __restrict__ gi,
           const float* __restrict__ W, const float* __restrict__ b,
           const float* __restrict__ addb, float* __restrict__ out, int R) {
  int row = blockIdx.x * 8 + (threadIdx.x >> 5);
  if (row >= R) return;
  int lane = threadIdx.x & 31;
  float av[KD / 32];
  if (GATHER) {
    int ri = gi[row], ci = gi[E2N + row];
#pragma unroll
    for (int i = 0; i < KD / 32; ++i) {
      int k = lane + 32 * i;
      av[i] = (k < HID) ? A[(size_t)ri * HID + k] : A[(size_t)ci * HID + (k - HID)];
    }
  } else {
#pragma unroll
    for (int i = 0; i < KD / 32; ++i) av[i] = A[(size_t)row * KD + lane + 32 * i];
  }
#pragma unroll
  for (int o = 0; o < OD; ++o) {
    float acc = 0.f;
#pragma unroll
    for (int i = 0; i < KD / 32; ++i) acc += av[i] * W[(size_t)(lane + 32 * i) * OD + o];
#pragma unroll
    for (int mm = 16; mm >= 1; mm >>= 1) acc += __shfl_xor(acc, mm, 32);
    if (lane == 0)
      out[(size_t)row * OD + o] = acc + b[o] + (ADDB ? addb[(size_t)row * OD + o] : 0.f);
  }
}

// ------------------------------ host side ----------------------------------
extern "C" void kernel_launch(void* const* d_in, const int* in_sizes, int n_in,
                              void* d_out, int out_size, void* d_ws, size_t ws_size,
                              hipStream_t stream) {
  (void)in_sizes; (void)n_in; (void)out_size; (void)ws_size;

  const float* x    = (const float*)d_in[0];
  const float* pos  = (const float*)d_in[1];
  const int*   ei2  = (const int*)d_in[2];
  const float* ea   = (const float*)d_in[3];
  const int*   ei3  = (const int*)d_in[5];

  int pi = 6;
  auto F = [&](int i) { return (const float*)d_in[i]; };
  const float *atomW = F(pi++), *atomB = F(pi++);
  const float *bondW = F(pi++), *bondB = F(pi++);
  const float *embW  = F(pi++), *embB  = F(pi++);
  struct EgP { const float *attW,*attB,*c1W,*c1B,*c2W,*c2B,*e1W,*e1B,*e2W,*e2B,*lnG,*lnB,*n1W,*n1B,*n2W,*n2B; } eg[NL];
  for (int i = 0; i < NL; ++i) {
    eg[i].attW = nullptr; eg[i].attB = nullptr;
    if (i > 0) { eg[i].attW = F(pi++); eg[i].attB = F(pi++); }
    eg[i].c1W = F(pi++); eg[i].c1B = F(pi++);
    eg[i].c2W = F(pi++); eg[i].c2B = F(pi++);
    eg[i].e1W = F(pi++); eg[i].e1B = F(pi++);
    eg[i].e2W = F(pi++); eg[i].e2B = F(pi++);
    eg[i].lnG = F(pi++); eg[i].lnB = F(pi++);
    eg[i].n1W = F(pi++); eg[i].n1B = F(pi++);
    eg[i].n2W = F(pi++); eg[i].n2B = F(pi++);
  }
  const float *fus1W = F(pi++), *fus1B = F(pi++);
  const float *fus2W = F(pi++), *fus2B = F(pi++);
  struct G2P { const float *eW,*eB,*linW,*linB,*lnG,*lnB; } g2[NL];
  for (int i = 0; i < NL; ++i) {
    g2[i].eW = F(pi++); g2[i].eB = F(pi++);
    g2[i].linW = F(pi++); g2[i].linB = F(pi++);
    g2[i].lnG = F(pi++); g2[i].lnB = F(pi++);
  }
  const float *posW = F(pi++), *posB = F(pi++);

  // workspace carve-up
  const size_t NB = (size_t)NN * HID;
  float* ws  = (float*)d_ws;
  float* nb0 = ws;             // xt / Ha / hn
  float* nb1 = nb0 + NB;       // out2d / Hb
  float* nb2 = nb1 + NB;       // agg
  float* nb3 = nb2 + NB;       // n1 out
  float* h2  = nb3 + NB;
  float* h3  = h2 + NB;
  float* pA  = h3 + NB;
  float* pB  = pA + (size_t)NN * 3;
  float* zb  = pB + (size_t)NN * 3;             // 256-float zero bias
  _Float16* m16 = (_Float16*)(zb + HID);
  _Float16* t16 = m16 + (size_t)CHUNK * HID;
  _Float16* wtG  = t16 + (size_t)CHUNK * HID;   // generic Wt (up to 512x256)
  _Float16* wtE2 = wtG  + (size_t)512 * HID;    // e2 Wt (per layer)
  _Float16* wtC1 = wtE2 + (size_t)512 * HID;    // c1 Wt (per layer)

  // output regions (atom[N,11], pos[N,3], bond[E2,4], hf[N,256])
  float* o_atom = (float*)d_out;
  float* o_pos  = o_atom + (size_t)NN * 11;
  float* o_bond = o_pos + (size_t)NN * 3;
  float* hf     = o_bond + (size_t)E2N * 4;

  const dim3 blk(256);
  const dim3 gN((NN + BM - 1) / BM, 2);

  // 0) zero bias buffer (used by the bias-free Ha/Hb GEMMs)
  k_zero<<<1, blk, 0, stream>>>(zb, HID);

  // 1) embed: h2 = x@W8 + b ; h3 = h2
  k_embed<<<NN, blk, 0, stream>>>(x, embW, embB, h2);
  hipMemcpyAsync(h3, h2, NB * sizeof(float), hipMemcpyDeviceToDevice, stream);

  // 2) 2D topology GNN
  for (int l = 0; l < NL; ++l) {
    k_cvtW<<<HID, blk, 0, stream>>>(g2[l].linW, wtG, HID);
    k_gemm<SRC_F32, EPI_NONE, OUT_F32><<<gN, blk, 0, stream>>>(
        h2, nullptr, nullptr, nullptr, nullptr, nullptr, nullptr, nullptr,
        nullptr, nullptr, wtG, g2[l].linB, nb0, nullptr, NN, HID, 0);
    hipMemcpyAsync(nb1, nb0, NB * sizeof(float), hipMemcpyDeviceToDevice, stream);
    k_scatter2d<<<E2N, blk, 0, stream>>>(nb0, nb1, ei2, ea, g2[l].eW, g2[l].eB);
    k_layernorm<0><<<NN, blk, 0, stream>>>(nb1, h2, g2[l].lnG, g2[l].lnB);
  }

  // 3) 3D EGNN
  hipMemcpyAsync(pA, pos, (size_t)NN * 3 * sizeof(float), hipMemcpyDeviceToDevice, stream);
  float* pcur = pA;
  float* pnxt = pB;
  for (int i = 0; i < NL; ++i) {
    // Ha = h3 @ We1[0:256], Hb = h3 @ We1[256:512]  (bias fused into gather)
    k_cvtW<<<HID, blk, 0, stream>>>(eg[i].e1W, wtG, HID);
    k_gemm<SRC_F32, EPI_NONE, OUT_F32><<<gN, blk, 0, stream>>>(
        h3, nullptr, nullptr, nullptr, nullptr, nullptr, nullptr, nullptr,
        nullptr, nullptr, wtG, zb, nb0, nullptr, NN, HID, 0);
    k_cvtW<<<HID, blk, 0, stream>>>(eg[i].e1W + (size_t)HID * HID, wtG, HID);
    k_gemm<SRC_F32, EPI_NONE, OUT_F32><<<gN, blk, 0, stream>>>(
        h3, nullptr, nullptr, nullptr, nullptr, nullptr, nullptr, nullptr,
        nullptr, nullptr, wtG, zb, nb1, nullptr, NN, HID, 0);
    hipMemcpyAsync(pnxt, pcur, (size_t)NN * 3 * sizeof(float), hipMemcpyDeviceToDevice, stream);
    k_zero<<<(int)((NB + 255) / 256), blk, 0, stream>>>(nb2, NB);
    k_cvtW<<<HID, blk, 0, stream>>>(eg[i].e2W, wtE2, HID);
    k_cvtW<<<HID, blk, 0, stream>>>(eg[i].c1W, wtC1, HID);

    for (int base = 0; base < E3N; base += CHUNK) {
      int cs = (E3N - base < CHUNK) ? (E3N - base) : CHUNK;
      dim3 gE(cs / BM, 2);
      // m = silu(e2(silu(Ha[row]+Hb[col]+rn*wr+b1)))  -> f16
      k_gemm<SRC_GATHER, EPI_SILU, OUT_F16><<<gE, blk, 0, stream>>>(
          nullptr, nullptr, nullptr, ei3, ei3 + E3N, nb0, nb1, pcur,
          eg[i].e1W + (size_t)2 * HID * HID, eg[i].e1B,
          wtE2, eg[i].e2B, nullptr, m16, cs, HID, base);
      // t = silu(c1(m)) -> f16
      k_gemm<SRC_F16, EPI_SILU, OUT_F16><<<gE, blk, 0, stream>>>(
          nullptr, nullptr, m16, nullptr, nullptr, nullptr, nullptr, nullptr,
          nullptr, nullptr, wtC1, eg[i].c1B, nullptr, t16, cs, HID, 0);
      // cd = c2(t) (+tanh last layer); pos scatter into pnxt
      k_cdpos<<<cs / 8, blk, 0, stream>>>(t16, eg[i].c2W, eg[i].c2B, ei3, ei3 + E3N,
                                          pcur, pnxt, base, cs, (i == NL - 1) ? 1 : 0);
      // agg += (sigmoid(att(m)))*m
      if (i > 0)
        k_aggatt<1><<<cs, blk, 0, stream>>>(m16, eg[i].attW, eg[i].attB, ei3, nb2, base);
      else
        k_aggatt<0><<<cs, blk, 0, stream>>>(m16, nullptr, nullptr, ei3, nb2, base);
    }
    // hn = n2(silu(n1(concat(h3, agg))));  h3 = LN(h3 + hn)
    k_cvtW<<<2 * HID, blk, 0, stream>>>(eg[i].n1W, wtG, 2 * HID);
    k_gemm<SRC_CONCAT, EPI_SILU, OUT_F32><<<gN, blk, 0, stream>>>(
        h3, nb2, nullptr, nullptr, nullptr, nullptr, nullptr, nullptr,
        nullptr, nullptr, wtG, eg[i].n1B, nb3, nullptr, NN, 2 * HID, 0);
    k_cvtW<<<HID, blk, 0, stream>>>(eg[i].n2W, wtG, HID);
    k_gemm<SRC_F32, EPI_NONE, OUT_F32><<<gN, blk, 0, stream>>>(
        nb3, nullptr, nullptr, nullptr, nullptr, nullptr, nullptr, nullptr,
        nullptr, nullptr, wtG, eg[i].n2B, nb0, nullptr, NN, HID, 0);
    k_layernorm<1><<<NN, blk, 0, stream>>>(nb0, h3, eg[i].lnG, eg[i].lnB);
    float* t = pcur; pcur = pnxt; pnxt = t;
  }

  // 4) fusion: hf = fus2(silu(fus1(concat(h2, h3))))
  k_cvtW<<<2 * HID, blk, 0, stream>>>(fus1W, wtG, 2 * HID);
  k_gemm<SRC_CONCAT, EPI_SILU, OUT_F32><<<gN, blk, 0, stream>>>(
      h2, h3, nullptr, nullptr, nullptr, nullptr, nullptr, nullptr,
      nullptr, nullptr, wtG, fus1B, nb3, nullptr, NN, 2 * HID, 0);
  k_cvtW<<<HID, blk, 0, stream>>>(fus2W, wtG, HID);
  k_gemm<SRC_F32, EPI_NONE, OUT_F32><<<gN, blk, 0, stream>>>(
      nb3, nullptr, nullptr, nullptr, nullptr, nullptr, nullptr, nullptr,
      nullptr, nullptr, wtG, fus2B, hf, nullptr, NN, HID, 0);

  // 5) heads
  k_smallout<HID, 11, 0, 0><<<(NN + 7) / 8, blk, 0, stream>>>(
      hf, nullptr, atomW, atomB, nullptr, o_atom, NN);
  k_smallout<HID, 3, 0, 1><<<(NN + 7) / 8, blk, 0, stream>>>(
      hf, nullptr, posW, posB, pcur, o_pos, NN);
  k_smallout<2 * HID, 4, 1, 0><<<(E2N + 7) / 8, blk, 0, stream>>>(
      hf, ei2, bondW, bondB, nullptr, o_bond, E2N);
}
